// TPWithWeights_44925357916441
// MI455X (gfx1250) — matrix operand order repaired and checked by reading, not compiled, not run on hardware
//
#include <hip/hip_runtime.h>
#include <hip/hip_bf16.h>

typedef __attribute__((ext_vector_type(16))) __bf16 v16bf;
typedef __attribute__((ext_vector_type(8)))  float  v8f;

// ---------------- static configuration ----------------
constexpr int NPATH = 11;
constexpr int PI1[NPATH] = {0,0,0,1,1,1,1,2,2,2,2};
constexpr int PI2[NPATH] = {0,1,2,0,1,1,2,0,1,2,2};
constexpr int PIO[NPATH] = {0,1,2,1,0,2,1,2,1,0,2};
constexpr int CGOFF[NPATH] = {0,1,10,35,44,53,98,143,168,213,238};
constexpr int CG_TOTAL = 363;
constexpr int OFF1C[3] = {0,64,256};
constexpr int OFF2C[3] = {0,1,4};
constexpr int OFFOC[3] = {0,64,256};
constexpr int DIM1 = 576;
constexpr int DIM2 = 9;
constexpr int NB   = 32768;
constexpr int XSTR = 580;                 // padded LDS row stride (floats)
constexpr int WAVES = 4;
constexpr int SMEM_FLOATS = 512 + WAVES * 16 * XSTR;
constexpr int SMEM_BYTES  = SMEM_FLOATS * 4;          // 150528 B
constexpr int WS_W_BYTE_OFF = 4096;

// ---------------- bf16 packing (native converts; RNE) ----------------
static __device__ __forceinline__ unsigned int pack2bf(float lo, float hi) {
    union { __bf16 b[2]; unsigned int u; } c;
    c.b[0] = (__bf16)lo;
    c.b[1] = (__bf16)hi;
    return c.u;
}

union ATile { v16bf v; unsigned int u[8]; };
union BTile { v16bf v; uint4 q[2]; unsigned int u[8]; };

// ---------------- Kernel 0: CG tables (fp64, one lane) ----------------
struct cplx { double re, im; };
static __device__ __forceinline__ cplx cmul(cplx a, cplx b) {
    return { a.re*b.re - a.im*b.im, a.re*b.im + a.im*b.re };
}
static __device__ __forceinline__ double factd(int n) {
    const double f[9] = {1.,1.,2.,6.,24.,120.,720.,5040.,40320.};
    return f[n];
}
static __device__ __forceinline__ int imaxi(int a,int b){return a>b?a:b;}
static __device__ __forceinline__ int imini(int a,int b){return a<b?a:b;}

static __device__ double su2cg(int j1,int m1,int j2,int m2,int j3,int m3) {
    if (m3 != m1 + m2) return 0.0;
    int vmin = imaxi(imaxi(-j1 + j2 + m3, -j1 + m1), 0);
    int vmax = imini(imini(j2 + j3 + m1, j3 - j1 + j2), j3 + m3);
    double C = sqrt((2.0*j3 + 1.0) * factd(j3+j1-j2) * factd(j3-j1+j2) * factd(j1+j2-j3)
                    * factd(j3+m3) * factd(j3-m3)
                    / (factd(j1+j2+j3+1) * factd(j1-m1) * factd(j1+m1)
                       * factd(j2-m2) * factd(j2+m2)));
    double S = 0.0;
    for (int v = vmin; v <= vmax; ++v) {
        double sgn = ((v + j2 + m2) & 1) ? -1.0 : 1.0;
        S += sgn * factd(j2+j3+m1-v) * factd(j1-m1+v)
             / (factd(v) * factd(j3-j1+j2-v) * factd(j3+m3-v) * factd(v+j1-j2-m3));
    }
    return C * S;
}

static __device__ void qbasis(int l, cplx q[5][5]) {
    const double s2 = 0.70710678118654752440;
    for (int a = 0; a < 5; ++a) for (int b = 0; b < 5; ++b) q[a][b] = {0.0, 0.0};
    for (int m = -l; m < 0; ++m) {
        q[l+m][l-m] = { s2, 0.0 };
        q[l+m][l+m] = { 0.0, -s2 };
    }
    q[l][l] = { 1.0, 0.0 };
    for (int m = 1; m <= l; ++m) {
        double sg = (m & 1) ? -1.0 : 1.0;
        q[l+m][l+m] = { sg * s2, 0.0 };
        q[l+m][l-m] = { 0.0, sg * s2 };
    }
    cplx ph;
    if      (l == 0) ph = { 1.0, 0.0 };
    else if (l == 1) ph = { 0.0, -1.0 };
    else             ph = { -1.0, 0.0 };
    for (int a = 0; a < 5; ++a) for (int b = 0; b < 5; ++b) q[a][b] = cmul(q[a][b], ph);
}

__global__ void cg_init_kernel(float* __restrict__ ws_cg) {
    if (threadIdx.x != 0 || blockIdx.x != 0) return;
    const double NRM[3] = {192.0, 256.0, 256.0};
    for (int p = 0; p < NPATH; ++p) {
        int l1 = PI1[p], l2 = PI2[p], lo = PIO[p];
        int d1 = 2*l1+1, d2 = 2*l2+1, d3 = 2*lo+1;
        double Csu[5][5][5];
        for (int a=0;a<5;++a) for(int b=0;b<5;++b) for(int c=0;c<5;++c) Csu[a][b][c]=0.0;
        for (int m1 = -l1; m1 <= l1; ++m1)
            for (int m2 = -l2; m2 <= l2; ++m2) {
                int m3 = m1 + m2;
                if (m3 >= -lo && m3 <= lo)
                    Csu[l1+m1][l2+m2][lo+m3] = su2cg(l1,m1,l2,m2,lo,m3);
            }
        cplx Q1[5][5], Q2[5][5], Q3[5][5];
        qbasis(l1, Q1); qbasis(l2, Q2); qbasis(lo, Q3);
        double R[5][5][5];
        double nrm2 = 0.0;
        for (int a = 0; a < d1; ++a)
          for (int b = 0; b < d2; ++b)
            for (int c = 0; c < d3; ++c) {
                double sre = 0.0;
                for (int i = 0; i < d1; ++i)
                  for (int k = 0; k < d2; ++k)
                    for (int n = 0; n < d3; ++n) {
                        double wv = Csu[i][k][n];
                        if (wv == 0.0) continue;
                        cplx t = cmul(Q1[i][a], Q2[k][b]);
                        cplx q3c = { Q3[n][c].re, -Q3[n][c].im };
                        t = cmul(t, q3c);
                        sre += t.re * wv;
                    }
                R[a][b][c] = sre;
                nrm2 += sre * sre;
            }
        double scale = sqrt((2.0*lo + 1.0) / NRM[lo]) / sqrt(nrm2);
        for (int c = 0; c < d3; ++c)
          for (int a = 0; a < d1; ++a)
            for (int b = 0; b < d2; ++b)
                ws_cg[CGOFF[p] + (c*d1 + a)*d2 + b] = (float)(R[a][b][c] * scale);
    }
}

// ---------------- Kernel 1: pack weights into WMMA-B bf16 tiles ----------------
__global__ void pack_w_kernel(const float* __restrict__ w, unsigned int* __restrict__ dst) {
    int tid = blockIdx.x * blockDim.x + threadIdx.x;
    if (tid >= NPATH * 2 * 4 * 256) return;
    int t    = tid >> 8;
    int rem  = tid & 255;
    int lane = rem >> 3;
    int r    = rem & 7;
    int wc   = t & 3;
    int puc  = t >> 2;
    int uc   = puc & 1;
    int p    = puc >> 1;
    int K    = ((lane >> 4) << 3) + ((r >= 4) ? 16 : 0) + 2*(r & 3);
    int u    = uc*32 + K;
    int wcol = wc*16 + (lane & 15);
    const float* wp = w + p * 4096;
    dst[tid] = pack2bf(wp[u*64 + wcol], wp[(u+1)*64 + wcol]);
}

// ---------------- Main fused kernel ----------------
template<int P, int LO, int K0, int K1>
static __device__ __forceinline__ void do_path(
    const float* __restrict__ cgs,
    const float* __restrict__ xrow,     // lane's LDS x1 row base
    const float (&x2r)[9],
    const uint4* __restrict__ wb,       // wpk + lane*2
    int hi,                             // lane >> 4
    v8f (&acc)[K1 - K0][4])
{
    constexpr int i1 = PI1[P], i2 = PI2[P];
    constexpr int D1 = 2*i1 + 1, D2 = 2*i2 + 1;
    constexpr int NK = K1 - K0;
    const float* xb  = xrow + OFF1C[i1] + hi * (8 * D1);  // single base VGPR
    const float* cgp = cgs + CGOFF[P];

    // M[k][i] = sum_j cg[i,j,k] * x2[j]
    float M[NK][D1];
    #pragma unroll
    for (int k = K0; k < K1; ++k)
        #pragma unroll
        for (int i = 0; i < D1; ++i) {
            float s = 0.f;
            #pragma unroll
            for (int j = 0; j < D2; ++j)
                s += cgp[(k*D1 + i)*D2 + j] * x2r[OFF2C[i2] + j];
            M[k - K0][i] = s;
        }

    #pragma unroll
    for (int uc = 0; uc < 2; ++uc) {
        BTile B[4];
        #pragma unroll
        for (int wc = 0; wc < 4; ++wc) {
            const int tile = (P*2 + uc)*4 + wc;        // constexpr after unroll
            B[wc].q[0] = wb[tile*64 + 0];
            B[wc].q[1] = wb[tile*64 + 1];
        }
        #pragma unroll
        for (int k = K0; k < K1; ++k) {
            ATile A;
            #pragma unroll
            for (int r = 0; r < 8; ++r) {
                const int koff = ((r >= 4) ? 16 : 0) + 2*(r & 3);
                // u0 even -> (uc*32+koff)*D1 even -> float2 8B-aligned
                const float2* xp = (const float2*)(xb + (uc*32 + koff) * D1);
                float y0 = 0.f, y1 = 0.f;
                #pragma unroll
                for (int t = 0; t < D1; ++t) {          // 2*D1 floats as D1 float2
                    float2 f = xp[t];
                    const int e0 = 2*t, e1 = 2*t + 1;
                    if (e0 < D1) y0 += M[k - K0][e0] * f.x;
                    else         y1 += M[k - K0][e0 - D1] * f.x;
                    if (e1 < D1) y0 += M[k - K0][e1] * f.y;
                    else         y1 += M[k - K0][e1 - D1] * f.y;
                }
                A.u[r] = pack2bf(y0, y1);
            }
            #pragma unroll
            for (int wc = 0; wc < 4; ++wc)
                acc[k - K0][wc] = __builtin_amdgcn_wmma_f32_16x16x32_bf16(
                    false, A.v, false, B[wc].v, (short)0, acc[k - K0][wc], false, false);
        }
    }
}

template<int LO, int K0, int K1>
static __device__ __forceinline__ void store_io(
    float* __restrict__ out, int n0, int lane, v8f (&acc)[K1 - K0][4])
{
    constexpr int DK = 2*LO + 1;
    const int col = lane & 15, hi = lane >> 4;
    float* base = out + (size_t)(n0 + hi*8) * DIM1 + OFFOC[LO] + col*DK;
    #pragma unroll
    for (int wc = 0; wc < 4; ++wc)
        #pragma unroll
        for (int r = 0; r < 8; ++r)
            #pragma unroll
            for (int k = K0; k < K1; ++k)
                base[r*DIM1 + wc*16*DK + k] = acc[k - K0][wc][r];
}

template<int NK>
static __device__ __forceinline__ void zero_acc(v8f (&acc)[NK][4]) {
    const v8f vz = {0.f,0.f,0.f,0.f,0.f,0.f,0.f,0.f};
    #pragma unroll
    for (int k = 0; k < NK; ++k)
        #pragma unroll
        for (int wc = 0; wc < 4; ++wc) acc[k][wc] = vz;
}

__global__ void __launch_bounds__(WAVES * 32)
tp_main_kernel(const float* __restrict__ x1,
               const float* __restrict__ x2,
               const float* __restrict__ ws_cg,
               const uint4* __restrict__ wpk,
               float* __restrict__ out)
{
    extern __shared__ float smem[];
    float* cgs = smem;
    const int lane = threadIdx.x & 31;
    const int wave = threadIdx.x >> 5;
    float* xs = smem + 512 + wave * (16 * XSTR);
    const int n0 = (blockIdx.x * WAVES + wave) * 16;

    for (int t = threadIdx.x; t < CG_TOTAL; t += blockDim.x) cgs[t] = ws_cg[t];

    {   // coalesced fp32 x1 tile -> LDS (float4 never straddles a row)
        const float4* src = (const float4*)(x1 + (size_t)n0 * DIM1);
        #pragma unroll
        for (int t = 0; t < (16 * DIM1) / (4 * 32); ++t) {   // 72 iters
            int idx  = t * 32 + lane;
            int elem = idx * 4;
            int row  = elem / DIM1;
            int colf = elem - row * DIM1;
            float4 v = src[idx];
            *(float4*)&xs[row * XSTR + colf] = v;
        }
    }

    const int m = lane & 15;
    const int hi = lane >> 4;
    float x2r[9];
    {
        const float* p2 = x2 + (size_t)(n0 + m) * DIM2;
        #pragma unroll
        for (int j = 0; j < 9; ++j) x2r[j] = p2[j];
    }

    __syncthreads();

    const float* xrow = xs + m * XSTR;
    const uint4* wb = wpk + lane * 2;

    // ---- io = 0 (lo=0): paths 0,4,9 ----
    {
        v8f acc[1][4];
        zero_acc<1>(acc);
        do_path<0,0,0,1>(cgs, xrow, x2r, wb, hi, acc);
        do_path<4,0,0,1>(cgs, xrow, x2r, wb, hi, acc);
        do_path<9,0,0,1>(cgs, xrow, x2r, wb, hi, acc);
        store_io<0,0,1>(out, n0, lane, acc);
    }
    // ---- io = 1 (lo=1): paths 1,3,6,8 ; k passes [0,2), [2,3) ----
    {
        v8f acc[2][4];
        zero_acc<2>(acc);
        do_path<1,1,0,2>(cgs, xrow, x2r, wb, hi, acc);
        do_path<3,1,0,2>(cgs, xrow, x2r, wb, hi, acc);
        do_path<6,1,0,2>(cgs, xrow, x2r, wb, hi, acc);
        do_path<8,1,0,2>(cgs, xrow, x2r, wb, hi, acc);
        store_io<1,0,2>(out, n0, lane, acc);
    }
    {
        v8f acc[1][4];
        zero_acc<1>(acc);
        do_path<1,1,2,3>(cgs, xrow, x2r, wb, hi, acc);
        do_path<3,1,2,3>(cgs, xrow, x2r, wb, hi, acc);
        do_path<6,1,2,3>(cgs, xrow, x2r, wb, hi, acc);
        do_path<8,1,2,3>(cgs, xrow, x2r, wb, hi, acc);
        store_io<1,2,3>(out, n0, lane, acc);
    }
    // ---- io = 2 (lo=2): paths 2,5,7,10 ; k passes [0,2), [2,4), [4,5) ----
    {
        v8f acc[2][4];
        zero_acc<2>(acc);
        do_path<2,2,0,2>(cgs, xrow, x2r, wb, hi, acc);
        do_path<5,2,0,2>(cgs, xrow, x2r, wb, hi, acc);
        do_path<7,2,0,2>(cgs, xrow, x2r, wb, hi, acc);
        do_path<10,2,0,2>(cgs, xrow, x2r, wb, hi, acc);
        store_io<2,0,2>(out, n0, lane, acc);
    }
    {
        v8f acc[2][4];
        zero_acc<2>(acc);
        do_path<2,2,2,4>(cgs, xrow, x2r, wb, hi, acc);
        do_path<5,2,2,4>(cgs, xrow, x2r, wb, hi, acc);
        do_path<7,2,2,4>(cgs, xrow, x2r, wb, hi, acc);
        do_path<10,2,2,4>(cgs, xrow, x2r, wb, hi, acc);
        store_io<2,2,4>(out, n0, lane, acc);
    }
    {
        v8f acc[1][4];
        zero_acc<1>(acc);
        do_path<2,2,4,5>(cgs, xrow, x2r, wb, hi, acc);
        do_path<5,2,4,5>(cgs, xrow, x2r, wb, hi, acc);
        do_path<7,2,4,5>(cgs, xrow, x2r, wb, hi, acc);
        do_path<10,2,4,5>(cgs, xrow, x2r, wb, hi, acc);
        store_io<2,4,5>(out, n0, lane, acc);
    }
}

extern "C" void kernel_launch(void* const* d_in, const int* in_sizes, int n_in,
                              void* d_out, int out_size, void* d_ws, size_t ws_size,
                              hipStream_t stream) {
    (void)in_sizes; (void)n_in; (void)out_size; (void)ws_size;
    const float* x1 = (const float*)d_in[0];
    const float* x2 = (const float*)d_in[1];
    const float* w  = (const float*)d_in[2];
    float* out = (float*)d_out;

    float* ws_cg = (float*)d_ws;
    unsigned int* ws_w = (unsigned int*)((char*)d_ws + WS_W_BYTE_OFF);

    (void)hipFuncSetAttribute(reinterpret_cast<const void*>(tp_main_kernel),
                              hipFuncAttributeMaxDynamicSharedMemorySize, SMEM_BYTES);

    cg_init_kernel<<<1, 32, 0, stream>>>(ws_cg);
    pack_w_kernel<<<(NPATH * 2 * 4 * 256) / 256, 256, 0, stream>>>(w, ws_w);
    tp_main_kernel<<<NB / (WAVES * 16), WAVES * 32, SMEM_BYTES, stream>>>(
        x1, x2, ws_cg, (const uint4*)ws_w, out);
}
